// RaggedEGCN_44573170597919
// MI455X (gfx1250) — matrix operand order repaired and checked by reading, not compiled, not run on hardware
//
#include <hip/hip_runtime.h>
#include <hip/hip_bf16.h>

#define K_NB 16
#define F_DIM 64
#define H_DIM 128
#define BIGF 1.0e9f

typedef __bf16 v16bf __attribute__((ext_vector_type(16)));
typedef float  v8f   __attribute__((ext_vector_type(8)));
typedef unsigned int u32x4 __attribute__((ext_vector_type(4)));

union FragU { u32x4 u[2]; v16bf v; };

__device__ __forceinline__ float selu_f(float x) {
    const float scale = 1.0507009873554805f;
    const float alpha = 1.6732632423543772f;
    return scale * (x > 0.f ? x : alpha * (__expf(x) - 1.f));
}

// ---------------------------------------------------------------------------
// KNN: one block per query point. Stage d2 row in LDS, 17 argmin passes.
// ---------------------------------------------------------------------------
__global__ void knn_kernel(const float* __restrict__ coords, const int* __restrict__ rs,
                           int* __restrict__ nidx_out, float* __restrict__ dist_out, int n) {
    __shared__ float sdist[5120];
    __shared__ float rv[256];
    __shared__ int   ri[256];
    int i = blockIdx.x;
    if (i >= n) return;
    int tid = threadIdx.x;
    int r0 = rs[0], r1 = rs[1], r2 = rs[2];
    int begin = (i < r1) ? r0 : r1;
    int end   = (i < r1) ? r1 : r2;
    int count = end - begin;
    if (count > 5120) count = 5120;
    float cx = coords[i*3+0], cy = coords[i*3+1], cz = coords[i*3+2];
    for (int t = tid; t < count; t += 256) {
        int c = begin + t;
        float dx = cx - coords[c*3+0];
        float dy = cy - coords[c*3+1];
        float dz = cz - coords[c*3+2];
        float d2 = dx*dx + dy*dy + dz*dz;
        sdist[t] = (c == i) ? -1.0f : d2;   // force self to be selected first
    }
    __syncthreads();
    for (int sel = 0; sel <= K_NB; ++sel) {
        float bv = BIGF; int bi = -1;
        for (int t = tid; t < count; t += 256) {
            float v = sdist[t];
            if (v < bv) { bv = v; bi = t; }
        }
        rv[tid] = bv; ri[tid] = bi;
        __syncthreads();
        for (int s = 128; s > 0; s >>= 1) {
            if (tid < s) {
                if (rv[tid+s] < rv[tid]) { rv[tid] = rv[tid+s]; ri[tid] = ri[tid+s]; }
            }
            __syncthreads();
        }
        if (tid == 0) {
            int widx = ri[0];
            float wval = rv[0];
            if (widx >= 0) sdist[widx] = BIGF;
            if (sel >= 1) {
                nidx_out[i*K_NB + sel - 1] = begin + widx;
                dist_out[i*K_NB + sel - 1] = (wval < 0.f) ? 0.f : wval;
            }
        }
        __syncthreads();
    }
}

// ---------------------------------------------------------------------------
// Weight prep: W[Kin][Nout] f32 -> Bt[Nout][Kpad] bf16 (transposed, padded)
// ---------------------------------------------------------------------------
__global__ void transpose_pad_kernel(const float* __restrict__ W, __bf16* __restrict__ Bt,
                                     int Kin, int Kpad, int Nout) {
    int id = blockIdx.x * 256 + threadIdx.x;
    int total = Nout * Kpad;
    if (id >= total) return;
    int nn = id / Kpad;
    int kk = id - nn * Kpad;
    float v = (kk < Kin) ? W[kk * Nout + nn] : 0.f;
    Bt[id] = (__bf16)v;
}

// ---------------------------------------------------------------------------
// WMMA GEMM: Out[M x 128] = act(A[M x Kd] * Bt^T + bias)
// Block = 256 threads = 8 waves; block owns 32 rows (2 M-subtiles), each wave
// one 16-col N-tile with two accumulators sharing each B fragment.
// Both WMMAs run unconditionally (garbage-safe); only staging loads and
// epilogue stores are guarded by M, as uniform branches OUTSIDE the K-loop.
// mode 0: A is a bf16 matrix (A tile staged via async global->LDS DMA).
// mode 1: gather edge rows [d | h_i | h_j | 0pad] (Kd == 160).
// ---------------------------------------------------------------------------
__global__ void gemm_wmma_kernel(const __bf16* __restrict__ A, const __bf16* __restrict__ Bt,
                                 const float* __restrict__ bias, float* __restrict__ outF,
                                 __bf16* __restrict__ outB, int M, int Kd, int applySelu, int mode,
                                 const float* __restrict__ h, const int* __restrict__ nidx,
                                 const float* __restrict__ dist) {
    __shared__ __align__(16) __bf16 As[32 * 192];
    int tid = threadIdx.x;
    int m0 = blockIdx.x * 32;

    if (mode == 0) {
        // Async DMA staging: memory -> LDS without VGPR round-trip (ASYNCcnt).
        int chunks = Kd >> 3;              // 16-byte chunks per row
        int total = 32 * chunks;
        for (int t = tid; t < total; t += 256) {
            int r = t / chunks, c = t - r * chunks;
            if (m0 + r < M) {
                unsigned lds_off = (unsigned)(size_t)(&As[r * Kd + c * 8]);
                unsigned long long gaddr =
                    (unsigned long long)(A + (size_t)(m0 + r) * Kd + c * 8);
                asm volatile("global_load_async_to_lds_b128 %0, %1, off"
                             :: "v"(lds_off), "v"(gaddr) : "memory");
            }
        }
        asm volatile("s_wait_asynccnt 0x0" ::: "memory");
    } else {
        // edge gather: cols [0]=d2, [1..64]=h[i], [65..128]=h[j], rest 0
        for (int t = tid; t < 32 * 160; t += 256) {
            int r = t / 160, c = t - r * 160;
            int m = m0 + r;
            int i = m >> 4;               // K_NB == 16
            float v;
            if (c == 0)            v = dist[m];
            else if (c < 65)       v = h[i * F_DIM + (c - 1)];
            else if (c < 129)      { int j = nidx[m]; v = h[j * F_DIM + (c - 65)]; }
            else                   v = 0.f;
            As[r * 160 + c] = (__bf16)v;
        }
    }
    __syncthreads();

    int lane = tid & 31;
    int wave = tid >> 5;
    int n0 = wave * 16;
    int rr = lane & 15;        // A row within tile / B+C column within tile
    int hh = lane >> 4;        // half-wave
    int nCol = n0 + rr;

    const u32x4* Brow  = (const u32x4*)(Bt + (size_t)nCol * Kd);
    __builtin_prefetch(Bt + (size_t)nCol * Kd, 0, 0);
    const u32x4* Arow0 = (const u32x4*)(As + rr * Kd);
    const u32x4* Arow1 = (const u32x4*)(As + (rr + 16) * Kd);

    v8f acc0 = {}, acc1 = {};
    int steps = Kd >> 5;       // K-step of 32
    for (int s = 0; s < steps; ++s) {
        int q = s * 4;         // u32x4 index base for this K-step
        FragU fb;
        // B 32x16 bf16 layout: lane = column, K = {32s+8hh .. +7, 32s+16+8hh .. +7}
        fb.u[0] = Brow[q + hh];
        fb.u[1] = Brow[q + 2 + hh];
        FragU fa0, fa1;
        // A 16x32 bf16 layout: lane half hh gets K = {32s+8hh .. +7, 32s+16+8hh .. +7}
        fa0.u[0] = Arow0[q + hh];
        fa0.u[1] = Arow0[q + 2 + hh];
        fa1.u[0] = Arow1[q + hh];
        fa1.u[1] = Arow1[q + 2 + hh];
        acc0 = __builtin_amdgcn_wmma_f32_16x16x32_bf16(false, fa0.v, false, fb.v,
                                                       (short)0, acc0, false, false);
        acc1 = __builtin_amdgcn_wmma_f32_16x16x32_bf16(false, fa1.v, false, fb.v,
                                                       (short)0, acc1, false, false);
    }

    float bn = bias[nCol];
#pragma unroll
    for (int v = 0; v < 8; ++v) {
        int m = m0 + v + 8 * hh;            // C/D layout: VGPR v holds row v (+8 upper half)
        float val = acc0[v] + bn;
        if (applySelu) val = selu_f(val);
        if (outF) outF[(size_t)m * H_DIM + nCol] = val;
        else      outB[(size_t)m * H_DIM + nCol] = (__bf16)val;
    }
    if (m0 + 16 < M) {                      // uniform branch, outside the K-loop
#pragma unroll
        for (int v = 0; v < 8; ++v) {
            int m = m0 + 16 + v + 8 * hh;
            float val = acc1[v] + bn;
            if (applySelu) val = selu_f(val);
            if (outF) outF[(size_t)m * H_DIM + nCol] = val;
            else      outB[(size_t)m * H_DIM + nCol] = (__bf16)val;
        }
    }
}

// ---------------------------------------------------------------------------
// w[m] = dot(C[m,:], W_c2) + b_c2
// ---------------------------------------------------------------------------
__global__ void wdot_kernel(const __bf16* __restrict__ C, const float* __restrict__ Wc2,
                            const float* __restrict__ bc2, float* __restrict__ w, int M) {
    int m = blockIdx.x * 256 + threadIdx.x;
    if (m >= M) return;
    const __bf16* row = C + (size_t)m * H_DIM;
    float acc = 0.f;
#pragma unroll 8
    for (int t = 0; t < H_DIM; ++t) acc += (float)row[t] * Wc2[t];
    w[m] = acc + bc2[0];
}

// ---------------------------------------------------------------------------
// coords_new[i] = coords[i] + mean_k (coords[i]-coords[j_k]) * w[i,k]
// ---------------------------------------------------------------------------
__global__ void trans_kernel(const float* __restrict__ coords, const int* __restrict__ nidx,
                             const float* __restrict__ w, float* __restrict__ coords_new, int n) {
    int i = blockIdx.x * 256 + threadIdx.x;
    if (i >= n) return;
    float cx = coords[i*3+0], cy = coords[i*3+1], cz = coords[i*3+2];
    float ax = 0.f, ay = 0.f, az = 0.f;
    for (int k = 0; k < K_NB; ++k) {
        int j = nidx[i*K_NB + k];
        float wk = w[i*K_NB + k];
        ax += (cx - coords[j*3+0]) * wk;
        ay += (cy - coords[j*3+1]) * wk;
        az += (cz - coords[j*3+2]) * wk;
    }
    const float inv = 1.0f / (float)K_NB;
    coords_new[i*3+0] = cx + ax * inv;
    coords_new[i*3+1] = cy + ay * inv;
    coords_new[i*3+2] = cz + az * inv;
}

// ---------------------------------------------------------------------------
// Agg[i, 0:128] = sum_k E[i*K+k, :]; Agg[i, 128:192] = h[i, :]
// ---------------------------------------------------------------------------
__global__ void esum_agg_kernel(const __bf16* __restrict__ E, const float* __restrict__ h,
                                __bf16* __restrict__ Agg, int n) {
    int id = blockIdx.x * 256 + threadIdx.x;
    int total = n * 192;
    if (id >= total) return;
    int i = id / 192;
    int c = id - i * 192;
    float v;
    if (c < H_DIM) {
        float s = 0.f;
        const __bf16* base = E + (size_t)(i * K_NB) * H_DIM + c;
#pragma unroll
        for (int k = 0; k < K_NB; ++k) s += (float)base[(size_t)k * H_DIM];
        v = s;
    } else {
        v = h[i * F_DIM + (c - H_DIM)];
    }
    Agg[id] = (__bf16)v;
}

// ---------------------------------------------------------------------------
extern "C" void kernel_launch(void* const* d_in, const int* in_sizes, int n_in,
                              void* d_out, int out_size, void* d_ws, size_t ws_size,
                              hipStream_t stream) {
    const float* h      = (const float*)d_in[0];
    const float* coords = (const float*)d_in[1];
    const int*   rs     = (const int*)d_in[2];
    const float* W_e1 = (const float*)d_in[3];
    const float* b_e1 = (const float*)d_in[4];
    const float* W_e2 = (const float*)d_in[5];
    const float* b_e2 = (const float*)d_in[6];
    const float* W_c1 = (const float*)d_in[7];
    const float* b_c1 = (const float*)d_in[8];
    const float* W_c2 = (const float*)d_in[9];
    const float* b_c2 = (const float*)d_in[10];
    const float* W_n1 = (const float*)d_in[11];
    const float* b_n1 = (const float*)d_in[12];
    const float* W_n2 = (const float*)d_in[13];
    const float* b_n2 = (const float*)d_in[14];

    const int n = in_sizes[1] / 3;     // N points
    const int M = n * K_NB;            // edges

    // d_out layout: out [N*H] f32 | coords_new [N*3] f32 | nidx [N*K] int | d [N*K] f32
    float* out_feats  = (float*)d_out;
    float* out_coords = out_feats + (size_t)n * H_DIM;
    int*   out_nidx   = (int*)(out_coords + (size_t)n * 3);
    float* out_d      = (float*)(out_nidx + (size_t)n * K_NB);

    // workspace carve
    char* p = (char*)d_ws;
    auto alloc = [&](size_t bytes) -> char* {
        char* r = p; p += (bytes + 255) & ~(size_t)255; return r;
    };
    __bf16* bt_e1 = (__bf16*)alloc((size_t)128 * 160 * 2);
    __bf16* bt_e2 = (__bf16*)alloc((size_t)128 * 128 * 2);
    __bf16* bt_c1 = (__bf16*)alloc((size_t)128 * 128 * 2);
    __bf16* bt_n1 = (__bf16*)alloc((size_t)128 * 192 * 2);
    __bf16* bt_n2 = (__bf16*)alloc((size_t)128 * 128 * 2);
    __bf16* E1    = (__bf16*)alloc((size_t)M * H_DIM * 2);
    __bf16* E     = (__bf16*)alloc((size_t)M * H_DIM * 2);
    float*  wv    = (float*)alloc((size_t)M * 4);
    __bf16* Agg   = (__bf16*)alloc((size_t)n * 192 * 2);
    __bf16* A1    = (__bf16*)alloc((size_t)n * H_DIM * 2);
    __bf16* Cbuf  = E1;   // safe reuse: G3 reads E, writes over E1

    const int gEdge = M / 32;                 // 5000 for N=10000
    const int gNode = (n + 31) / 32;          // 313

    // 1. weight prep (bf16 transposed+padded)
    transpose_pad_kernel<<<(128*160 + 255)/256, 256, 0, stream>>>(W_e1, bt_e1, 129, 160, 128);
    transpose_pad_kernel<<<(128*128 + 255)/256, 256, 0, stream>>>(W_e2, bt_e2, 128, 128, 128);
    transpose_pad_kernel<<<(128*128 + 255)/256, 256, 0, stream>>>(W_c1, bt_c1, 128, 128, 128);
    transpose_pad_kernel<<<(128*192 + 255)/256, 256, 0, stream>>>(W_n1, bt_n1, 192, 192, 128);
    transpose_pad_kernel<<<(128*128 + 255)/256, 256, 0, stream>>>(W_n2, bt_n2, 128, 128, 128);

    // 2. KNN
    knn_kernel<<<n, 256, 0, stream>>>(coords, rs, out_nidx, out_d, n);

    // 3. edge MLP (WMMA bf16), gather fused into first GEMM
    gemm_wmma_kernel<<<gEdge, 256, 0, stream>>>(nullptr, bt_e1, b_e1, nullptr, E1,
                                                M, 160, 1, 1, h, out_nidx, out_d);
    gemm_wmma_kernel<<<gEdge, 256, 0, stream>>>(E1, bt_e2, b_e2, nullptr, E,
                                                M, 128, 1, 0, nullptr, nullptr, nullptr);
    gemm_wmma_kernel<<<gEdge, 256, 0, stream>>>(E, bt_c1, b_c1, nullptr, Cbuf,
                                                M, 128, 1, 0, nullptr, nullptr, nullptr);

    // 4. edge weights + coordinate update
    wdot_kernel<<<(M + 255)/256, 256, 0, stream>>>(Cbuf, W_c2, b_c2, wv, M);
    trans_kernel<<<(n + 255)/256, 256, 0, stream>>>(coords, out_nidx, wv, out_coords, n);

    // 5. aggregation + node MLP (WMMA bf16)
    esum_agg_kernel<<<(n*192 + 255)/256, 256, 0, stream>>>(E, h, Agg, n);
    gemm_wmma_kernel<<<gNode, 256, 0, stream>>>(Agg, bt_n1, b_n1, nullptr, A1,
                                                n, 192, 1, 0, nullptr, nullptr, nullptr);
    gemm_wmma_kernel<<<gNode, 256, 0, stream>>>(A1, bt_n2, b_n2, out_feats, nullptr,
                                                n, 128, 0, 0, nullptr, nullptr, nullptr);
}
